// NetworkGNN_22634477650042
// MI455X (gfx1250) — matrix operand order repaired and checked by reading, not compiled, not run on hardware
//
#include <hip/hip_runtime.h>
#include <hip/hip_bf16.h>

// NetworkGNN forward for MI455X (gfx1250, wave32, WMMA).
// Dense GEMMs: v_wmma_f32_16x16x32_f16 (f16 in, f32 accumulate).
// A slab staged via GLOBAL_LOAD_ASYNC_TO_LDS_B128 (ASYNCcnt) -> ds_load_b128 fragments.
// conv+lin GEMMs fused (shared A fragments, 8 WMMA/wave) with the GCN epilogue.
// Sparse aggregation: one wave per edge, float4 gather + fp32 atomics (roofline stage).
// Entire working set (~105 MB) fits in the 192 MB L2 -> L2-resident pipeline.

#define HID      128
#define NLAYERS  3
#define NGRAPHS  128
#define LDS_STRIDE 136   // halves per row: 272 B -> 68 dwords -> distinct LDS banks per row

typedef _Float16 v16h __attribute__((ext_vector_type(16)));
typedef _Float16 h8   __attribute__((ext_vector_type(8)));
typedef float    v8f  __attribute__((ext_vector_type(8)));

// ---------------------------------------------------------------- preprocessing

__global__ void k_prep_edges(const long long* __restrict__ e64,
                             int* __restrict__ src, int* __restrict__ dst,
                             float* __restrict__ deg, int E) {
    int i = blockIdx.x * blockDim.x + threadIdx.x;
    if (i >= E) return;
    int s = (int)e64[i];
    int d = (int)e64[(size_t)E + i];
    src[i] = s;
    dst[i] = d;
    atomicAdd(&deg[d], 1.0f);
}

__global__ void k_dinv(const float* __restrict__ deg, float* __restrict__ dinv,
                       float* __restrict__ selfn, int N) {
    int i = blockIdx.x * blockDim.x + threadIdx.x;
    if (i >= N) return;
    float di = rsqrtf(deg[i] + 1.0f);   // +1 for self loop
    dinv[i]  = di;
    selfn[i] = di * di;
}

__global__ void k_enorm(const int* __restrict__ src, const int* __restrict__ dst,
                        const float* __restrict__ dinv, float* __restrict__ enorm, int E) {
    int i = blockIdx.x * blockDim.x + threadIdx.x;
    if (i >= E) return;
    enorm[i] = dinv[src[i]] * dinv[dst[i]];
}

__global__ void k_init_h(const long long* __restrict__ x_idx, const float* __restrict__ emb,
                         float* __restrict__ acc, _Float16* __restrict__ accH, int NH) {
    int i = blockIdx.x * blockDim.x + threadIdx.x;
    if (i >= NH) return;
    int n = i >> 7, k = i & 127;
    float v = emb[((int)x_idx[n]) * HID + k];
    acc[i]  = v;
    accH[i] = (_Float16)v;
}

// Pack one fp32 [K=128][N=128] weight matrix into the per-lane WMMA B-fragment
// layout (16-bit B 32x16 tiles): out index = ((kt*8 + nt)*32 + lane)*16 + j,
// where j maps to K = (j&7) + 8*(lane>>4) + 16*(j>>3) + 32*kt, N = nt*16 + (lane&15).
__global__ void k_pack_w(const float* __restrict__ W, _Float16* __restrict__ Bp) {
    int i = blockIdx.x * blockDim.x + threadIdx.x;   // 0 .. 4095
    if (i >= 4096) return;
    int j    = i & 15;
    int lane = (i >> 4) & 31;
    int nt   = (i >> 9) & 7;
    int kt   = (i >> 12) & 3;
    int v = j >> 1, r = j & 1;
    int kB = ((2 * v + r) & 7) + 8 * (lane >> 4) + 16 * ((2 * v + r) >> 3) + 32 * kt;
    int n  = nt * 16 + (lane & 15);
    Bp[i] = (_Float16)W[kB * HID + n];
}

__global__ void k_f32_to_f16(const float* __restrict__ in, _Float16* __restrict__ out, int n) {
    int i = blockIdx.x * blockDim.x + threadIdx.x;
    if (i >= n) return;
    out[i] = (_Float16)in[i];
}

// ---------------------------------------------------------------- WMMA GEMM core
// Block = 256 threads = 8 waves = one 16-row slab of A; wave nt -> 16-col tile nt.
// A slab staged into LDS with async global->LDS B128 copies (no VGPR round-trip,
// tracked by ASYNCcnt), padded rows to dodge bank conflicts; fragments read back
// with ds_load_b128; K=128 -> 4 chained v_wmma_f32_16x16x32_f16 per output tile.

__device__ __forceinline__ void stage_A(const _Float16* __restrict__ A, _Float16* sA,
                                        int row0, int M) {
    int t     = threadIdx.x;
    int row   = t >> 4;          // 0..15
    int chunk = t & 15;          // 16 B chunk within the 256 B row
    int rs    = row0 + row;
    rs = (rs >= M) ? (M - 1) : rs;                 // clamp without EXEC divergence
    // LDS aperture rule (ISA 10.2): low 32 bits of a generic LDS pointer are the
    // LDS byte offset -> feed directly as the async-load VDST address.
    unsigned lds_off = (unsigned)(uintptr_t)(sA + row * LDS_STRIDE + chunk * 8);
    unsigned long long ga =
        (unsigned long long)(uintptr_t)(A + (size_t)rs * HID + chunk * 8);
    asm volatile("global_load_async_to_lds_b128 %0, %1, off"
                 :: "v"(lds_off), "v"(ga) : "memory");
    asm volatile("s_wait_asynccnt 0x0" ::: "memory");
    __syncthreads();
}

__device__ __forceinline__ void load_a_frags(const _Float16* sA, int m, int halfSel,
                                             v16h a[4]) {
    const _Float16* Arow = sA + m * LDS_STRIDE;
#pragma unroll
    for (int kt = 0; kt < 4; ++kt) {
        // 16-bit A 16x32 fragment: j=0..7 -> K = kt*32 + halfSel*8 + j
        //                          j=8..15 -> K = kt*32 + halfSel*8 + 16 + (j-8)
        h8 lo = *(const h8*)(Arow + kt * 32 + halfSel * 8);
        h8 hi = *(const h8*)(Arow + kt * 32 + halfSel * 8 + 16);
#pragma unroll
        for (int p = 0; p < 8; ++p) { a[kt][p] = lo[p]; a[kt][8 + p] = hi[p]; }
    }
}

__device__ __forceinline__ v8f wmma_k128(const v16h a[4], const _Float16* __restrict__ Bp,
                                         int nt, int lane) {
    v8f c = {};
#pragma unroll
    for (int kt = 0; kt < 4; ++kt) {
        v16h b = *(const v16h*)(Bp + ((size_t)((kt * 8 + nt) * 32 + lane)) * 16);
        c = __builtin_amdgcn_wmma_f32_16x16x32_f16(
                false, a[kt], false, b, (short)0, c, false, false);
    }
    return c;
}

// Fused conv+lin GEMM with GCN epilogue:
//   xw  = acc @ convW                        (needed by the edge gather)
//   agg = acc @ linW + xw*selfn + cb + lb    (self-loop term + biases pre-folded)
__global__ void k_gemm_dual(const _Float16* __restrict__ A,
                            const _Float16* __restrict__ Bc, const _Float16* __restrict__ Bl,
                            const float* __restrict__ selfn,
                            const float* __restrict__ cb, const float* __restrict__ lb,
                            float* __restrict__ xw, float* __restrict__ agg, int M) {
    __shared__ _Float16 sA[16 * LDS_STRIDE];
    int row0    = blockIdx.x << 4;
    int lane    = threadIdx.x & 31;
    int nt      = threadIdx.x >> 5;
    int halfSel = lane >> 4;
    int m       = lane & 15;

    stage_A(A, sA, row0, M);
    v16h a[4];
    load_a_frags(sA, m, halfSel, a);

    v8f cc = wmma_k128(a, Bc, nt, lane);
    v8f cl = wmma_k128(a, Bl, nt, lane);

    int   col  = nt * 16 + m;
    float bias = cb[col] + lb[col];
    if (row0 + 16 <= M) {                       // wave-uniform fast path
#pragma unroll
        for (int v = 0; v < 8; ++v) {
            int   row = row0 + v + 8 * halfSel;
            float xv  = cc[v];
            size_t o  = (size_t)row * HID + col;
            xw[o]  = xv;
            agg[o] = cl[v] + xv * selfn[row] + bias;
        }
    } else {
#pragma unroll
        for (int v = 0; v < 8; ++v) {
            int row = row0 + v + 8 * halfSel;
            if (row < M) {
                float xv = cc[v];
                size_t o = (size_t)row * HID + col;
                xw[o]  = xv;
                agg[o] = cl[v] + xv * selfn[row] + bias;
            }
        }
    }
}

// last_mlp GEMM fused with bias + elu + global_add_pool (atomic scatter by graph id).
__global__ void k_gemm_mlp_pool(const _Float16* __restrict__ A, const _Float16* __restrict__ Bp,
                                const float* __restrict__ mb,
                                const long long* __restrict__ batch,
                                float* __restrict__ pooled, int M) {
    __shared__ _Float16 sA[16 * LDS_STRIDE];
    int row0    = blockIdx.x << 4;
    int lane    = threadIdx.x & 31;
    int nt      = threadIdx.x >> 5;
    int halfSel = lane >> 4;
    int m       = lane & 15;

    stage_A(A, sA, row0, M);
    v16h a[4];
    load_a_frags(sA, m, halfSel, a);
    v8f c = wmma_k128(a, Bp, nt, lane);

    int   col  = nt * 16 + m;
    float bias = mb[col];
#pragma unroll
    for (int v = 0; v < 8; ++v) {
        int row = row0 + v + 8 * halfSel;
        if (row < M) {
            float val = c[v] + bias;
            val = (val > 0.0f) ? val : (expf(val) - 1.0f);   // elu
            int g = (int)batch[row];
            atomicAdd(&pooled[g * HID + col], val);
        }
    }
}

// prediction head GEMM fused with bias + /10, writes straight to d_out.
__global__ void k_gemm_out(const _Float16* __restrict__ A, const _Float16* __restrict__ Bp,
                           const float* __restrict__ pb, float* __restrict__ out, int M) {
    __shared__ _Float16 sA[16 * LDS_STRIDE];
    int row0    = blockIdx.x << 4;
    int lane    = threadIdx.x & 31;
    int nt      = threadIdx.x >> 5;
    int halfSel = lane >> 4;
    int m       = lane & 15;

    stage_A(A, sA, row0, M);
    v16h a[4];
    load_a_frags(sA, m, halfSel, a);
    v8f c = wmma_k128(a, Bp, nt, lane);

    int   col  = nt * 16 + m;
    float bias = pb[col];
#pragma unroll
    for (int v = 0; v < 8; ++v) {
        int row = row0 + v + 8 * halfSel;
        out[(size_t)row * HID + col] = (c[v] + bias) * 0.1f;
    }
}

// ---------------------------------------------------------------- sparse + pointwise

// One wave per edge: lane handles 4 features (32 lanes * 4 = 128).
__global__ void k_edge_scatter(const int* __restrict__ src, const int* __restrict__ dst,
                               const float* __restrict__ enorm, const float* __restrict__ xw,
                               float* __restrict__ agg, int E) {
    int gw   = (blockIdx.x * blockDim.x + threadIdx.x) >> 5;   // global wave = edge
    int lane = threadIdx.x & 31;
    if (gw >= E) return;
    int   s  = src[gw];
    int   d  = dst[gw];
    float en = enorm[gw];
    float4 v = ((const float4*)(xw + (size_t)s * HID))[lane];
    float* ad = agg + (size_t)d * HID + lane * 4;
    atomicAdd(ad + 0, v.x * en);
    atomicAdd(ad + 1, v.y * en);
    atomicAdd(ad + 2, v.z * en);
    atomicAdd(ad + 3, v.w * en);
}

__global__ void k_combine(const float* __restrict__ agg, float* __restrict__ acc,
                          _Float16* __restrict__ accH, int NH) {
    int i = blockIdx.x * blockDim.x + threadIdx.x;
    if (i >= NH) return;
    float x = agg[i];
    x = (x > 0.0f) ? x : (expf(x) - 1.0f);   // elu
    float na = acc[i] + x;
    acc[i]  = na;
    accH[i] = (_Float16)na;
}

// ---------------------------------------------------------------- launcher

extern "C" void kernel_launch(void* const* d_in, const int* in_sizes, int n_in,
                              void* d_out, int out_size, void* d_ws, size_t ws_size,
                              hipStream_t stream) {
    const long long* x_idx  = (const long long*)d_in[0];
    const long long* eidx   = (const long long*)d_in[1];
    const long long* batch  = (const long long*)d_in[2];
    const float* node_emb   = (const float*)d_in[3];
    const float* conv_W     = (const float*)d_in[4];
    const float* conv_b     = (const float*)d_in[5];
    const float* lin_W      = (const float*)d_in[6];
    const float* lin_b      = (const float*)d_in[7];
    const float* mlp_W      = (const float*)d_in[8];
    const float* mlp_b      = (const float*)d_in[9];
    const float* pred_W     = (const float*)d_in[10];
    const float* pred_b     = (const float*)d_in[11];

    const int N  = in_sizes[0];
    const int E  = in_sizes[1] / 2;
    const int NH = N * HID;
    const int GH = NGRAPHS * HID;

    // Workspace carve-out (256B aligned), ~105 MB total.
    char*  base = (char*)d_ws;
    size_t off  = 0;
    auto take = [&](size_t bytes) -> void* {
        off = (off + 255) & ~(size_t)255;
        void* p = base + off;
        off += bytes;
        return p;
    };
    int*      src     = (int*)take((size_t)E * 4);
    int*      dst     = (int*)take((size_t)E * 4);
    float*    deg     = (float*)take((size_t)N * 4);
    float*    dinv    = (float*)take((size_t)N * 4);
    float*    selfn   = (float*)take((size_t)N * 4);
    float*    enorm   = (float*)take((size_t)E * 4);
    float*    acc     = (float*)take((size_t)NH * 4);
    _Float16* accH    = (_Float16*)take((size_t)NH * 2);
    float*    xw      = (float*)take((size_t)NH * 4);
    float*    agg     = (float*)take((size_t)NH * 4);
    _Float16* wpack   = (_Float16*)take((size_t)8 * 4096 * 2);  // conv0..2, lin0..2, mlp, pred
    float*    pooled  = (float*)take((size_t)GH * 4);
    _Float16* pooledH = (_Float16*)take((size_t)GH * 2);
    (void)ws_size; (void)n_in; (void)out_size;

    const int T = 256;
    hipMemsetAsync(deg,    0, (size_t)N  * 4, stream);
    hipMemsetAsync(pooled, 0, (size_t)GH * 4, stream);

    k_prep_edges<<<(E + T - 1) / T, T, 0, stream>>>(eidx, src, dst, deg, E);
    k_dinv      <<<(N + T - 1) / T, T, 0, stream>>>(deg, dinv, selfn, N);
    k_enorm     <<<(E + T - 1) / T, T, 0, stream>>>(src, dst, dinv, enorm, E);
    k_init_h    <<<(NH + T - 1) / T, T, 0, stream>>>(x_idx, node_emb, acc, accH, NH);

    for (int l = 0; l < NLAYERS; ++l) {
        k_pack_w<<<16, T, 0, stream>>>(conv_W + (size_t)l * HID * HID, wpack + (size_t)l * 4096);
        k_pack_w<<<16, T, 0, stream>>>(lin_W  + (size_t)l * HID * HID, wpack + (size_t)(3 + l) * 4096);
    }
    k_pack_w<<<16, T, 0, stream>>>(mlp_W,  wpack + (size_t)6 * 4096);
    k_pack_w<<<16, T, 0, stream>>>(pred_W, wpack + (size_t)7 * 4096);

    const int MB = (N + 15) / 16;   // 3125 row slabs
    for (int l = 0; l < NLAYERS; ++l) {
        k_gemm_dual<<<MB, T, 0, stream>>>(accH,
                                          wpack + (size_t)l * 4096,
                                          wpack + (size_t)(3 + l) * 4096,
                                          selfn,
                                          conv_b + (size_t)l * HID,
                                          lin_b + (size_t)l * HID,
                                          xw, agg, N);
        k_edge_scatter<<<(E + 7) / 8, T, 0, stream>>>(src, dst, enorm, xw, agg, E);
        k_combine<<<(NH + T - 1) / T, T, 0, stream>>>(agg, acc, accH, NH);
    }

    // last_mlp (+bias+elu) fused with global_add_pool
    k_gemm_mlp_pool<<<MB, T, 0, stream>>>(accH, wpack + (size_t)6 * 4096,
                                          mlp_b, batch, pooled, N);

    // prediction head: (pooled @ pred_W + pred_b) / 10 straight into d_out
    k_f32_to_f16<<<(GH + T - 1) / T, T, 0, stream>>>(pooled, pooledH, GH);
    k_gemm_out<<<NGRAPHS / 16, T, 0, stream>>>(pooledH, wpack + (size_t)7 * 4096,
                                               pred_b, (float*)d_out, NGRAPHS);
}
